// Lilly_82514911691023
// MI455X (gfx1250) — matrix operand order repaired and checked by reading, not compiled, not run on hardware
//
#include <hip/hip_runtime.h>
#include <hip/hip_bf16.h>

// ---------------- problem constants (match reference) ----------------
#define B_   4
#define S_   1024
#define D_   512
#define H_   8
#define HD_  64
#define L_   2
#define F_   2048
#define V_   32000
#define M_   (B_ * S_)          // 4096 token rows
#define QKVN (3 * D_)           // 1536

// NUM token stats: mean(0..999)=499.5, std=sqrt(83333.25)+1e-6
#define NUM_MEAN 499.5f
#define NUM_STD  288.6749913f
#define SQRT_D   22.627416998f  // sqrt(512)

typedef __attribute__((ext_vector_type(16))) _Float16 v16h;
typedef __attribute__((ext_vector_type(8)))  _Float16 v8h;
typedef __attribute__((ext_vector_type(8)))  float    v8f;

// =====================================================================
// 1) Embedding (numeric softsign powers OR table gather) + pos-enc + LN
//    one 256-thread block per (b,s) row
// =====================================================================
__global__ __launch_bounds__(256) void embed_ln_kernel(
    const int* __restrict__ src, const float* __restrict__ emb_table,
    const float* __restrict__ g, const float* __restrict__ beta,
    float* __restrict__ x)
{
  int row = blockIdx.x;          // b*S + s
  int s   = row & (S_ - 1);
  int tok = src[row];
  __shared__ float sh[D_];
  __shared__ float r1[256], r2[256];
  int tid = threadIdx.x;

  for (int d = tid; d < D_; d += 256) {
    float e;
    if (tok < 1000) {
      float n  = ((float)tok - NUM_MEAN) * (1.0f / NUM_STD);
      float ss = n / (1.0f + fabsf(n));
      float mag = powf(fabsf(ss), (float)(d + 1));
      e = (ss < 0.0f && ((d + 1) & 1)) ? -mag : mag;   // ss^(d+1)
    } else {
      e = emb_table[(size_t)tok * D_ + d];
    }
    int j = d >> 1;
    float ang = (float)s * __expf(-9.2103403720f * (float)(2 * j) * (1.0f / (float)D_));
    float p = (d & 1) ? cosf(ang) : sinf(ang);
    sh[d] = e * SQRT_D + p;
  }
  __syncthreads();

  float a  = sh[tid];
  float b2 = sh[tid + 256];
  r1[tid] = a + b2;
  r2[tid] = a * a + b2 * b2;
  __syncthreads();
  for (int off = 128; off > 0; off >>= 1) {
    if (tid < off) { r1[tid] += r1[tid + off]; r2[tid] += r2[tid + off]; }
    __syncthreads();
  }
  float mean = r1[0] * (1.0f / D_);
  float var  = r2[0] * (1.0f / D_) - mean * mean;
  float rstd = rsqrtf(var + 1e-5f);

  x[(size_t)row * D_ + tid]       = (a  - mean) * rstd * g[tid]       + beta[tid];
  x[(size_t)row * D_ + tid + 256] = (b2 - mean) * rstd * g[tid + 256] + beta[tid + 256];
}

// =====================================================================
// 2) Row LayerNorm (D=512), one 256-thread block per row
// =====================================================================
__global__ __launch_bounds__(256) void ln_kernel(
    const float* __restrict__ in, const float* __restrict__ g,
    const float* __restrict__ beta, float* __restrict__ out)
{
  int row = blockIdx.x;
  int tid = threadIdx.x;
  const float* xr = in + (size_t)row * D_;
  __shared__ float r1[256], r2[256];

  float a  = xr[tid];
  float b2 = xr[tid + 256];
  r1[tid] = a + b2;
  r2[tid] = a * a + b2 * b2;
  __syncthreads();
  for (int off = 128; off > 0; off >>= 1) {
    if (tid < off) { r1[tid] += r1[tid + off]; r2[tid] += r2[tid + off]; }
    __syncthreads();
  }
  float mean = r1[0] * (1.0f / D_);
  float var  = r2[0] * (1.0f / D_) - mean * mean;
  float rstd = rsqrtf(var + 1e-5f);

  out[(size_t)row * D_ + tid]       = (a  - mean) * rstd * g[tid]       + beta[tid];
  out[(size_t)row * D_ + tid + 256] = (b2 - mean) * rstd * g[tid + 256] + beta[tid + 256];
}

// =====================================================================
// 3) WMMA GEMM: Y[M,N] = A[M,K] @ W[N,K]^T + bias (+residual)(+ReLU)
//    block tile 128x128, 8 waves (4Mx2N) of 32x64, K-step 32
//    f32 -> f16 staged through LDS; fragments load as ds_load_b128.
//    Requires M%128==0, N%128==0, K%32==0 (true for all uses here).
//    Epilogue flags are template params -> straight-line code.
// =====================================================================
#define BM 128
#define BN 128
#define BK 32
#define LPITCH (BK + 8)   // 40 halves = 80B row pitch (16B aligned)

template <int HASRES, int HASRELU>
__global__ __launch_bounds__(256) void gemm_wmma_kernel(
    const float* __restrict__ A, const float* __restrict__ W,
    const float* __restrict__ bias, const float* __restrict__ residual,
    float* __restrict__ Y, int M, int N, int K)
{
  __shared__ __align__(16) _Float16 As[BM][LPITCH];
  __shared__ __align__(16) _Float16 Ws[BN][LPITCH];

  int bm = blockIdx.y * BM, bn = blockIdx.x * BN;
  int tid  = threadIdx.x;
  int wave = tid >> 5, lane = tid & 31;
  int wm = (wave & 3) * 32, wn = (wave >> 2) * 64;
  int l16 = lane & 15, half = lane >> 4;

  v8f acc[2][4] = {};

  for (int k0 = 0; k0 < K; k0 += BK) {
    // stage A tile (128x32 f32 -> f16): 1024 float4, 4 per thread, coalesced
#pragma unroll
    for (int i = 0; i < 4; i++) {
      int idx = tid + 256 * i;            // 0..1023
      int r = idx >> 3, c4 = idx & 7;
      const float* ap = A + (size_t)(bm + r) * K + k0;
      float4 v = ((const float4*)ap)[c4];
      if (k0 + BK < K) __builtin_prefetch(ap + BK + c4 * 4, 0, 1);
      _Float16* dst = &As[r][c4 * 4];
      dst[0] = (_Float16)v.x; dst[1] = (_Float16)v.y;
      dst[2] = (_Float16)v.z; dst[3] = (_Float16)v.w;
    }
    // stage W tile (128x32 f32 -> f16)
#pragma unroll
    for (int i = 0; i < 4; i++) {
      int idx = tid + 256 * i;            // 0..1023
      int r = idx >> 3, c4 = idx & 7;
      const float* wp = W + (size_t)(bn + r) * K + k0;
      float4 v = ((const float4*)wp)[c4];
      if (k0 + BK < K) __builtin_prefetch(wp + BK + c4 * 4, 0, 1);
      _Float16* dst = &Ws[r][c4 * 4];
      dst[0] = (_Float16)v.x; dst[1] = (_Float16)v.y;
      dst[2] = (_Float16)v.z; dst[3] = (_Float16)v.w;
    }
    __syncthreads();

#pragma unroll
    for (int i = 0; i < 2; i++) {
      // A fragment: m = l16 (+tile), k = (h<8 ? h : h+8) + 8*half
      const _Float16* arow = &As[wm + i * 16 + l16][0];
      v8h alo = *(const v8h*)(arow + 8 * half);
      v8h ahi = *(const v8h*)(arow + 16 + 8 * half);
      v16h a;
#pragma unroll
      for (int t = 0; t < 8; t++) { a[t] = alo[t]; a[t + 8] = ahi[t]; }

#pragma unroll
      for (int j = 0; j < 4; j++) {
        // B fragment (W^T tile): n = l16, k = (h<8 ? h : h+8) + 8*half
        const _Float16* wrow = &Ws[wn + j * 16 + l16][0];
        v8h blo = *(const v8h*)(wrow + 8 * half);
        v8h bhi = *(const v8h*)(wrow + 16 + 8 * half);
        v16h bf;
#pragma unroll
        for (int t = 0; t < 8; t++) { bf[t] = blo[t]; bf[t + 8] = bhi[t]; }

        acc[i][j] = __builtin_amdgcn_wmma_f32_16x16x32_f16(
            false, a, false, bf, (short)0, acc[i][j], false, false);
      }
    }
    __syncthreads();
  }

  // epilogue: C/D layout m = v + 8*half, n = l16  (straight-line, templated)
#pragma unroll
  for (int i = 0; i < 2; i++) {
#pragma unroll
    for (int j = 0; j < 4; j++) {
      int n  = bn + wn + j * 16 + l16;
      int m0 = bm + wm + i * 16 + 8 * half;
      float bi = bias[n];
      size_t off = (size_t)m0 * N + n;
#pragma unroll
      for (int v = 0; v < 8; v++) {
        float val = acc[i][j][v] + bi;
        if (HASRES) val += residual[off];
        if (HASRELU) val = fmaxf(val, 0.0f);
        Y[off] = val;
        off += (size_t)N;
      }
    }
  }
}

// =====================================================================
// 4) Causal flash attention, one wave per (b, head, 16-query tile)
//    qkv rows: [q(0..511) | k(512..1023) | v(1024..1535)], hd=64
// =====================================================================
__global__ __launch_bounds__(32) void attn_kernel(
    const float* __restrict__ qkv, float* __restrict__ o)
{
  int qb = blockIdx.x, h = blockIdx.y, b = blockIdx.z;
  int q0 = qb * 16;
  int lane = threadIdx.x, l16 = lane & 15, half = lane >> 4;

  __shared__ __align__(16) _Float16 Kls[16][72];   // [key][hd], pitch 144B
  __shared__ __align__(16) _Float16 Vt[HD_][24];   // [hd][key], pitch 48B
  __shared__ __align__(16) _Float16 Pls[16][24];   // [qrow][key]

  size_t base = (size_t)b * S_ * QKVN;

  // Q fragments (contraction over hd = 64 -> two K=32 chunks)
  const float* qrow = qkv + base + (size_t)(q0 + l16) * QKVN + h * HD_;
  v16h aQ[2];
#pragma unroll
  for (int c = 0; c < 2; c++) {
#pragma unroll
    for (int t = 0; t < 8; t++) {
      aQ[c][t]     = (_Float16)qrow[c * 32 + 8 * half + t];
      aQ[c][t + 8] = (_Float16)qrow[c * 32 + 16 + 8 * half + t];
    }
  }

  float m_i[8], l_i[8];
#pragma unroll
  for (int v = 0; v < 8; v++) { m_i[v] = -3.0e38f; l_i[v] = 0.0f; }
  v8f oacc[4] = {};

  int r = lane >> 1, ch = lane & 1;   // staging: row r, 32-col half ch

  for (int j = 0; j <= qb; j++) {
    int k0 = j * 16;
    const float* krow = qkv + base + (size_t)(k0 + r) * QKVN + D_     + h * HD_ + ch * 32;
    const float* vrow = qkv + base + (size_t)(k0 + r) * QKVN + 2 * D_ + h * HD_ + ch * 32;
#pragma unroll
    for (int t = 0; t < 32; t++) {
      Kls[r][ch * 32 + t] = (_Float16)krow[t];
      Vt[ch * 32 + t][r]  = (_Float16)vrow[t];
    }
    __syncthreads();

    // S = Q K^T  (B frag: n=key=l16, k=hd = c*32 + 8*half + (h<8 ? h : h+8))
    v8f sacc = {};
#pragma unroll
    for (int c = 0; c < 2; c++) {
      const _Float16* kr = &Kls[l16][c * 32 + 8 * half];
      v8h blo = *(const v8h*)kr;
      v8h bhi = *(const v8h*)(kr + 16);
      v16h bK;
#pragma unroll
      for (int t = 0; t < 8; t++) { bK[t] = blo[t]; bK[t + 8] = bhi[t]; }
      sacc = __builtin_amdgcn_wmma_f32_16x16x32_f16(
          false, aQ[c], false, bK, (short)0, sacc, false, false);
    }

    // causal mask + online softmax (row = v + 8*half, col = l16)
    float p[8], sc[8];
#pragma unroll
    for (int v = 0; v < 8; v++) {
      float xv = sacc[v] * 0.125f;                 // 1/sqrt(64)
      int qr = q0 + v + 8 * half, kc = k0 + l16;
      if (kc > qr) xv = -3.0e38f;
      float rm = xv;
      rm = fmaxf(rm, __shfl_xor(rm, 1));
      rm = fmaxf(rm, __shfl_xor(rm, 2));
      rm = fmaxf(rm, __shfl_xor(rm, 4));
      rm = fmaxf(rm, __shfl_xor(rm, 8));
      float newm = fmaxf(m_i[v], rm);
      float pv = __expf(xv - newm);
      float rs = pv;
      rs += __shfl_xor(rs, 1); rs += __shfl_xor(rs, 2);
      rs += __shfl_xor(rs, 4); rs += __shfl_xor(rs, 8);
      float scv = __expf(m_i[v] - newm);
      l_i[v] = l_i[v] * scv + rs;
      m_i[v] = newm;
      p[v] = pv; sc[v] = scv;
    }
#pragma unroll
    for (int t = 0; t < 4; t++)
#pragma unroll
      for (int v = 0; v < 8; v++) oacc[t][v] *= sc[v];

    __syncthreads();                 // WAR on Pls from previous iter
#pragma unroll
    for (int v = 0; v < 8; v++) Pls[v + 8 * half][l16] = (_Float16)p[v];
    __syncthreads();

    // P as A fragment (keys 16..31 zero-padded)
    v8h plo = *(const v8h*)(&Pls[l16][8 * half]);
    v16h aP;
#pragma unroll
    for (int t = 0; t < 8; t++) { aP[t] = plo[t]; aP[t + 8] = (_Float16)0.0f; }

    // O += P V   (B frag: n = hd col, k = key = 8*half + h; h>=8 -> keys 16-31 = 0)
#pragma unroll
    for (int t = 0; t < 4; t++) {
      v16h bV;
#pragma unroll
      for (int tt = 0; tt < 16; tt++) bV[tt] = (_Float16)0.0f;
      {
        const _Float16* vr = &Vt[t * 16 + l16][8 * half];
        v8h vlo = *(const v8h*)vr;
#pragma unroll
        for (int tt = 0; tt < 8; tt++) { bV[tt] = vlo[tt]; }
      }
      oacc[t] = __builtin_amdgcn_wmma_f32_16x16x32_f16(
          false, aP, false, bV, (short)0, oacc[t], false, false);
    }
    __syncthreads();                 // WAR on Kls/Vt before next iter
  }

  // write O / l_i  (rows v + 8*half, cols h*64 + t*16 + l16)
#pragma unroll
  for (int t = 0; t < 4; t++)
#pragma unroll
    for (int v = 0; v < 8; v++) {
      int qr = q0 + v + 8 * half;
      o[((size_t)b * S_ + qr) * D_ + h * HD_ + t * 16 + l16] = oacc[t][v] / l_i[v];
    }
}

// =====================================================================
// launcher
// =====================================================================
extern "C" void kernel_launch(void* const* d_in, const int* in_sizes, int n_in,
                              void* d_out, int out_size, void* d_ws, size_t ws_size,
                              hipStream_t stream)
{
  (void)in_sizes; (void)n_in; (void)out_size; (void)ws_size;

  const int*   src        = (const int*)  d_in[0];
  const float* emb_table  = (const float*)d_in[1];
  const float* ln_emb_g   = (const float*)d_in[2];
  const float* ln_emb_b   = (const float*)d_in[3];
  const float* in_proj_w  = (const float*)d_in[4];
  const float* in_proj_b  = (const float*)d_in[5];
  const float* out_proj_w = (const float*)d_in[6];
  const float* out_proj_b = (const float*)d_in[7];
  const float* ln1_g      = (const float*)d_in[8];
  const float* ln1_b      = (const float*)d_in[9];
  const float* ln2_g      = (const float*)d_in[10];
  const float* ln2_b      = (const float*)d_in[11];
  const float* ffn_w1     = (const float*)d_in[12];
  const float* ffn_b1     = (const float*)d_in[13];
  const float* ffn_w2     = (const float*)d_in[14];
  const float* ffn_b2     = (const float*)d_in[15];
  const float* out_w      = (const float*)d_in[16];
  const float* out_b      = (const float*)d_in[17];
  float* logits = (float*)d_out;

  // workspace layout (bytes)
  char* ws = (char*)d_ws;
  float* x    = (float*)(ws);                                    //  8 MiB: [4096,512]
  float* hbuf = (float*)(ws + 8388608);                          //  8 MiB: [4096,512]
  float* qkv  = (float*)(ws + 2 * 8388608);                      // 24 MiB: [4096,1536]
  float* attn = (float*)(ws + 2 * 8388608 + 25165824);           //  8 MiB: [4096,512]
  float* ffnh = (float*)(ws + 3 * 8388608 + 25165824);           // 32 MiB: [4096,2048]

  embed_ln_kernel<<<M_, 256, 0, stream>>>(src, emb_table, ln_emb_g, ln_emb_b, x);

  for (int l = 0; l < L_; l++) {
    ln_kernel<<<M_, 256, 0, stream>>>(x, ln1_g + l * D_, ln1_b + l * D_, hbuf);

    gemm_wmma_kernel<0, 0><<<dim3(QKVN / BN, M_ / BM), 256, 0, stream>>>(
        hbuf, in_proj_w + (size_t)l * QKVN * D_, in_proj_b + l * QKVN,
        nullptr, qkv, M_, QKVN, D_);

    attn_kernel<<<dim3(S_ / 16, H_, B_), 32, 0, stream>>>(qkv, attn);

    gemm_wmma_kernel<1, 0><<<dim3(D_ / BN, M_ / BM), 256, 0, stream>>>(
        attn, out_proj_w + (size_t)l * D_ * D_, out_proj_b + l * D_,
        x, x, M_, D_, D_);

    ln_kernel<<<M_, 256, 0, stream>>>(x, ln2_g + l * D_, ln2_b + l * D_, hbuf);

    gemm_wmma_kernel<0, 1><<<dim3(F_ / BN, M_ / BM), 256, 0, stream>>>(
        hbuf, ffn_w1 + (size_t)l * F_ * D_, ffn_b1 + l * F_,
        nullptr, ffnh, M_, F_, D_);

    gemm_wmma_kernel<1, 0><<<dim3(D_ / BN, M_ / BM), 256, 0, stream>>>(
        ffnh, ffn_w2 + (size_t)l * D_ * F_, ffn_b2 + l * D_,
        x, x, M_, D_, F_);
  }

  gemm_wmma_kernel<0, 0><<<dim3(V_ / BN, M_ / BM), 256, 0, stream>>>(
      x, out_w, out_b, nullptr, logits, M_, V_, D_);
}
